// RigorousTropicalKAN_88098369175695
// MI455X (gfx1250) — compile-verified, matrix operand order checked by reading, not run on hardware
//
#include <hip/hip_runtime.h>

// Problem constants (from reference)
#define B_DIM   512
#define IN_DIM  256
#define OUT_DIM 256
#define NDEG    6

#define NBT   (B_DIM / 16)     // 32 b-tiles
#define NOTI  (OUT_DIM / 16)   // 16 o-tiles
#define NSEG  4                // i-dimension split for parallelism (deterministic 2-pass)
#define SEG_LEN (IN_DIM / NSEG) // 64
#define CHUNK 16               // i's per log-amortization chunk (product stays in f32 range)

typedef __attribute__((ext_vector_type(16))) _Float16 v16h;
typedef __attribute__((ext_vector_type(8)))  _Float16 v8h;
typedef __attribute__((ext_vector_type(8)))  float    v8f;

// ---------------------------------------------------------------------------
// Pack x -> A-fragments. SA[i][bt][lane][8 halves]:
//   lanes 0-15 (M = rows b): elements 0..7 = K=0..7 -> s_d = exp(d*x - max(0,5x)), d<6, else 0
//   lanes 16-31: zeros (K=8..15 / 24..31 unused)
// ---------------------------------------------------------------------------
__global__ void pack_x_kernel(const float* __restrict__ x, _Float16* __restrict__ SA) {
    int t = blockIdx.x * blockDim.x + threadIdx.x;   // ((i*NBT + bt)*32 + lane)
    if (t >= IN_DIM * NBT * 32) return;
    int lane = t & 31;
    int bt   = (t >> 5) & (NBT - 1);
    int i    = t >> 10;                              // / (32*NBT)
    float v[8] = {0.f,0.f,0.f,0.f,0.f,0.f,0.f,0.f};
    if (lane < 16) {
        int b = bt * 16 + lane;
        float xv = x[(size_t)b * IN_DIM + i];
        float m  = fmaxf(0.0f, 5.0f * xv);           // exact max_d(x*d); cancels in p-q
#pragma unroll
        for (int d = 0; d < NDEG; ++d) v[d] = __expf((float)d * xv - m);
    }
    _Float16* dst = SA + (size_t)t * 8;
#pragma unroll
    for (int d = 0; d < 8; ++d) dst[d] = (_Float16)v[d];
}

// ---------------------------------------------------------------------------
// Pack weights -> B-fragments. EP/EQ[i][ot][lane][8 halves]:
//   lanes 0-15 (N = cols o): elements 0..7 = K=0..7 -> exp(w[i,o,d]), d<6, else 0
//   lanes 16-31: zeros (K=8..15 / 24..31 unused)
// ---------------------------------------------------------------------------
__global__ void pack_w_kernel(const float* __restrict__ wp, const float* __restrict__ wq,
                              _Float16* __restrict__ EP, _Float16* __restrict__ EQ) {
    int t = blockIdx.x * blockDim.x + threadIdx.x;   // ((i*NOTI + ot)*32 + lane)
    if (t >= IN_DIM * NOTI * 32) return;
    int lane = t & 31;
    int ot   = (t >> 5) & (NOTI - 1);
    int i    = t >> 9;                               // / (32*NOTI)
    float p[8] = {0.f,0.f,0.f,0.f,0.f,0.f,0.f,0.f};
    float q[8] = {0.f,0.f,0.f,0.f,0.f,0.f,0.f,0.f};
    if (lane < 16) {
        int o = ot * 16 + lane;
        const float* wps = wp + ((size_t)i * OUT_DIM + o) * NDEG;
        const float* wqs = wq + ((size_t)i * OUT_DIM + o) * NDEG;
#pragma unroll
        for (int d = 0; d < NDEG; ++d) { p[d] = __expf(wps[d]); q[d] = __expf(wqs[d]); }
    }
    _Float16* dp = EP + (size_t)t * 8;
    _Float16* dq = EQ + (size_t)t * 8;
#pragma unroll
    for (int d = 0; d < 8; ++d) { dp[d] = (_Float16)p[d]; dq[d] = (_Float16)q[d]; }
}

// ---------------------------------------------------------------------------
// Main kernel: one wave per (o-tile, b-tile, i-segment).
// Per i: 2x v_wmma_f32_16x16x32_f16 (P and Q rank-6 dots, K zero-padded),
// elementwise product accumulation; one log2 pair per CHUNK=16 i's.
// ---------------------------------------------------------------------------
__global__ __launch_bounds__(32) void kan_wmma_kernel(const _Float16* __restrict__ SA,
                                                      const _Float16* __restrict__ EP,
                                                      const _Float16* __restrict__ EQ,
                                                      float* __restrict__ part) {
    const int ot   = blockIdx.x;       // 0..NOTI-1
    const int bt   = blockIdx.y;       // 0..NBT-1
    const int seg  = blockIdx.z;       // 0..NSEG-1
    const int lane = threadIdx.x;      // wave32

    const size_t astr = (size_t)NBT  * 32 * 8;  // per-i stride (halves)
    const size_t bstr = (size_t)NOTI * 32 * 8;

    const _Float16* ap = SA + (((size_t)(seg * SEG_LEN) * NBT  + bt) * 32 + lane) * 8;
    const _Float16* pp = EP + (((size_t)(seg * SEG_LEN) * NOTI + ot) * 32 + lane) * 8;
    const _Float16* qp = EQ + (((size_t)(seg * SEG_LEN) * NOTI + ot) * 32 + lane) * 8;

    const v8h  z8    = {};
    const v8f  czero = {};
    v8f acc = {};

    for (int ic = 0; ic < SEG_LEN; ic += CHUNK) {
        if (ic + CHUNK < SEG_LEN) {                 // prefetch next chunk (global_prefetch_b8)
            __builtin_prefetch(ap + CHUNK * astr, 0, 1);
            __builtin_prefetch(pp + CHUNK * bstr, 0, 1);
            __builtin_prefetch(qp + CHUNK * bstr, 0, 1);
        }
        v8f prodP = {1.f,1.f,1.f,1.f,1.f,1.f,1.f,1.f};
        v8f prodQ = {1.f,1.f,1.f,1.f,1.f,1.f,1.f,1.f};
        for (int j = 0; j < CHUNK; ++j) {
            v8h a8 = *(const v8h*)ap;
            v8h p8 = *(const v8h*)pp;
            v8h q8 = *(const v8h*)qp;
            ap += astr; pp += bstr; qp += bstr;
            // zero-extend K=0..7 fragment halves to the full K=32 operand
            v16h a  = __builtin_shufflevector(a8, z8, 0,1,2,3,4,5,6,7, 8,9,10,11,12,13,14,15);
            v16h bp = __builtin_shufflevector(p8, z8, 0,1,2,3,4,5,6,7, 8,9,10,11,12,13,14,15);
            v16h bq = __builtin_shufflevector(q8, z8, 0,1,2,3,4,5,6,7, 8,9,10,11,12,13,14,15);
            v8f dp = __builtin_amdgcn_wmma_f32_16x16x32_f16(false, a, false, bp,
                                                            (short)0, czero, false, false);
            v8f dq = __builtin_amdgcn_wmma_f32_16x16x32_f16(false, a, false, bq,
                                                            (short)0, czero, false, false);
            prodP *= dp;     // Π_i P_i  (range-safe for 16 factors)
            prodQ *= dq;     // Π_i Q_i
        }
#pragma unroll
        for (int e = 0; e < 8; ++e)
            acc[e] += __log2f(prodP[e]) - __log2f(prodQ[e]);
    }

    // C/D f32 layout: element e -> row M = e (lanes 0-15) / 8+e (lanes 16-31), col N = lane&15
    const int n     = lane & 15;
    const int mbase = (lane < 16) ? 0 : 8;
    float* dst = part + (size_t)seg * B_DIM * OUT_DIM
                      + ((size_t)(bt * 16 + mbase)) * OUT_DIM + (size_t)(ot * 16 + n);
#pragma unroll
    for (int e = 0; e < 8; ++e) dst[(size_t)e * OUT_DIM] = acc[e];
}

// ---------------------------------------------------------------------------
// Deterministic reduction over i-segments; convert log2 -> ln.
// ---------------------------------------------------------------------------
__global__ void reduce_kernel(const float* __restrict__ part, float* __restrict__ out) {
    int t = blockIdx.x * blockDim.x + threadIdx.x;
    if (t >= B_DIM * OUT_DIM) return;
    float s = 0.f;
#pragma unroll
    for (int g = 0; g < NSEG; ++g) s += part[(size_t)g * B_DIM * OUT_DIM + t];
    out[t] = s * 0.69314718055994530942f;   // ln2
}

extern "C" void kernel_launch(void* const* d_in, const int* in_sizes, int n_in,
                              void* d_out, int out_size, void* d_ws, size_t ws_size,
                              hipStream_t stream) {
    const float* x  = (const float*)d_in[0];
    // d_in[1] = slopes (arange(0..5), baked into pack_x_kernel)
    const float* wp = (const float*)d_in[2];
    const float* wq = (const float*)d_in[3];

    _Float16* SA = (_Float16*)d_ws;                                // 4 MB
    _Float16* EP = SA + (size_t)IN_DIM * NBT  * 32 * 8;            // 2 MB
    _Float16* EQ = EP + (size_t)IN_DIM * NOTI * 32 * 8;            // 2 MB
    float*  part = (float*)(EQ + (size_t)IN_DIM * NOTI * 32 * 8);  // 2 MB
    float*   out = (float*)d_out;

    int nx = IN_DIM * NBT * 32;
    pack_x_kernel<<<(nx + 255) / 256, 256, 0, stream>>>(x, SA);
    int nw = IN_DIM * NOTI * 32;
    pack_w_kernel<<<(nw + 255) / 256, 256, 0, stream>>>(wp, wq, EP, EQ);

    dim3 grid(NOTI, NBT, NSEG);          // 16 x 32 x 4 = 2048 waves
    kan_wmma_kernel<<<grid, 32, 0, stream>>>(SA, EP, EQ, part);

    int no = B_DIM * OUT_DIM;
    reduce_kernel<<<(no + 255) / 256, 256, 0, stream>>>(part, out);
}
